// NodeLevelGNN_16329465659829
// MI455X (gfx1250) — compile-verified, matrix operand order checked by reading, not run on hardware
//
#include <hip/hip_runtime.h>
#include <math.h>

#define N_NODES 10000
#define E_EDGES 320000
#define K_NEAR  30
#define NEG_SLOPE 0.2f

typedef __attribute__((ext_vector_type(16))) _Float16 v16h;
typedef __attribute__((ext_vector_type(8)))  _Float16 v8h;
typedef __attribute__((ext_vector_type(8)))  float    v8f;

// ---------------------------------------------------------------------------
// utility kernels
// ---------------------------------------------------------------------------
__global__ void k_fill(float* p, float v, int n) {
    int i = blockIdx.x * blockDim.x + threadIdx.x;
    if (i < n) p[i] = v;
}

__global__ void k_copy_x(float* feat, const float* x, int ldf) {
    int idx = blockIdx.x * blockDim.x + threadIdx.x;
    if (idx >= N_NODES * 10) return;
    int v = idx / 10, c = idx % 10;
    feat[v * ldf + c] = x[idx];
}

__global__ void k_bias_init(float* feat, const float* bias, int dout, int ldf, int coff) {
    int idx = blockIdx.x * blockDim.x + threadIdx.x;
    if (idx >= N_NODES * dout) return;
    int v = idx / dout, k = idx % dout;
    feat[v * ldf + coff + k] = bias[k];
}

__device__ __forceinline__ float atomicMaxF(float* addr, float val) {
    if (val >= 0.f)
        return __int_as_float(atomicMax((int*)addr, __float_as_int(val)));
    else
        return __uint_as_float(atomicMin((unsigned int*)addr, __float_as_uint(val)));
}

// ---------------------------------------------------------------------------
// f32 -> zero-padded f16 staging (branch-free WMMA operand fetch afterwards)
// ---------------------------------------------------------------------------
__global__ void k_cvt_a(const float* __restrict__ A, int lda, int offA,
                        int K, int Kpad, _Float16* __restrict__ out) {
    int idx = blockIdx.x * blockDim.x + threadIdx.x;
    if (idx >= N_NODES * Kpad) return;
    int r = idx / Kpad, k = idx % Kpad;
    out[idx] = (k < K) ? (_Float16)A[(size_t)r * lda + offA + k] : (_Float16)0.f;
}

__global__ void k_cvt_w(const float* __restrict__ W, int Ncols, int K,
                        int Npad, int Kpad, _Float16* __restrict__ out) {
    int idx = blockIdx.x * blockDim.x + threadIdx.x;
    if (idx >= Npad * Kpad) return;
    int n = idx / Kpad, k = idx % Kpad;
    out[idx] = (n < Ncols && k < K) ? (_Float16)W[(size_t)n * K + k] : (_Float16)0.f;
}

// ---------------------------------------------------------------------------
// WMMA GEMM on padded f16 operands:
//   C[m, offC+n] (+= bias[n]) = sum_k Ah[m,k] * Wh[n,k]
// Ah: [M x Kpad] f16 row-major (M multiple of 16), Wh: [Npad x Kpad] f16.
// One wave = one 16x16 tile; per K-chunk: 4x global_load_b128 + 1 WMMA.
// Lane layouts per CDNA5 ISA 7.12.2 (16-bit A 16x32; f32 C/D 16x16).
// ---------------------------------------------------------------------------
__global__ void k_gemm_wmma(const _Float16* __restrict__ Ah,
                            const _Float16* __restrict__ Wh,
                            const float* __restrict__ bias,
                            float* __restrict__ C, int ldc, int offC,
                            int Ncols, int Kpad) {
    int lane  = threadIdx.x & 31;
    int l16   = lane & 15;
    int khalf = (lane < 16) ? 0 : 8;

    const _Float16* arow = Ah + ((size_t)blockIdx.x * 16 + l16) * Kpad;
    const _Float16* brow = Wh + ((size_t)blockIdx.y * 16 + l16) * Kpad;

    v8f acc = {};
    for (int kc = 0; kc < Kpad; kc += 32) {
        v8h a0 = *(const v8h*)(arow + kc + khalf);
        v8h a1 = *(const v8h*)(arow + kc + 16 + khalf);
        v8h b0 = *(const v8h*)(brow + kc + khalf);
        v8h b1 = *(const v8h*)(brow + kc + 16 + khalf);
        v16h a = __builtin_shufflevector(a0, a1, 0, 1, 2, 3, 4, 5, 6, 7,
                                         8, 9, 10, 11, 12, 13, 14, 15);
        v16h b = __builtin_shufflevector(b0, b1, 0, 1, 2, 3, 4, 5, 6, 7,
                                         8, 9, 10, 11, 12, 13, 14, 15);
        acc = __builtin_amdgcn_wmma_f32_16x16x32_f16(false, a, false, b,
                                                     (short)0, acc, false, false);
    }

    int col = blockIdx.y * 16 + l16;
    if (col < Ncols) {
        float bv = (bias != nullptr) ? bias[col] : 0.f;
        int mbase = blockIdx.x * 16 + ((lane < 16) ? 0 : 8);
#pragma unroll
        for (int r = 0; r < 8; ++r)
            C[(size_t)(mbase + r) * ldc + offC + col] = acc[r] + bv;
    }
}

// ---------------------------------------------------------------------------
// GAT pieces
// ---------------------------------------------------------------------------
__global__ void k_scores(const float* __restrict__ h,
                         const float* __restrict__ att_s,
                         const float* __restrict__ att_d,
                         float* asrc, float* adst, int dout) {
    int v = blockIdx.x * blockDim.x + threadIdx.x;
    if (v >= N_NODES) return;
    float s = 0.f, d = 0.f;
    for (int k = 0; k < dout; ++k) {
        float hv = h[v * dout + k];
        s += hv * att_s[k];
        d += hv * att_d[k];
    }
    asrc[v] = s;
    adst[v] = d;
}

__global__ void k_ve(const float* w_e, const float* att_e, float* ve, int dout) {
    int t = threadIdx.x;
    if (t < 4) {
        float acc = 0.f;
        for (int d = 0; d < dout; ++d) acc += w_e[d * 4 + t] * att_e[d];
        ve[t] = acc;
    }
}

__global__ void k_ea_accum(const float* __restrict__ ea, float* acc4) {
    float l0 = 0.f, l1 = 0.f, l2 = 0.f, l3 = 0.f;
    for (int e = blockIdx.x * blockDim.x + threadIdx.x; e < E_EDGES;
         e += gridDim.x * blockDim.x) {
        l0 += ea[e * 4 + 0];
        l1 += ea[e * 4 + 1];
        l2 += ea[e * 4 + 2];
        l3 += ea[e * 4 + 3];
    }
    atomicAdd(&acc4[0], l0);
    atomicAdd(&acc4[1], l1);
    atomicAdd(&acc4[2], l2);
    atomicAdd(&acc4[3], l3);
}

__global__ void k_alpha(const int* __restrict__ ei,
                        const float* __restrict__ asrc,
                        const float* __restrict__ adst,
                        const float* __restrict__ eattr,
                        const float* __restrict__ ve,
                        const float* __restrict__ eamean, float invE, int hasEA,
                        float* alpha, float* amax) {
    int idx = blockIdx.x * blockDim.x + threadIdx.x;
    const int M = E_EDGES + N_NODES;
    if (idx >= M) return;
    int s, d;
    float et = 0.f;
    if (idx < E_EDGES) {
        s = ei[idx];
        d = ei[E_EDGES + idx];
        if (hasEA)
            et = eattr[idx * 4 + 0] * ve[0] + eattr[idx * 4 + 1] * ve[1] +
                 eattr[idx * 4 + 2] * ve[2] + eattr[idx * 4 + 3] * ve[3];
    } else {
        s = d = idx - E_EDGES;  // self loop
        if (hasEA)
            et = (eamean[0] * ve[0] + eamean[1] * ve[1] + eamean[2] * ve[2] +
                  eamean[3] * ve[3]) * invE;
    }
    float a = asrc[s] + adst[d] + et;
    a = (a >= 0.f) ? a : NEG_SLOPE * a;
    alpha[idx] = a;
    atomicMaxF(&amax[d], a);
}

__global__ void k_exp(const int* __restrict__ ei,
                      const float* __restrict__ alpha,
                      const float* __restrict__ amax,
                      float* evals, float* ssum) {
    int idx = blockIdx.x * blockDim.x + threadIdx.x;
    const int M = E_EDGES + N_NODES;
    if (idx >= M) return;
    int d = (idx < E_EDGES) ? ei[E_EDGES + idx] : idx - E_EDGES;
    float e = expf(alpha[idx] - amax[d]);
    evals[idx] = e;
    atomicAdd(&ssum[d], e);
}

__global__ void k_scatter(const int* __restrict__ ei,
                          const float* __restrict__ evals,
                          const float* __restrict__ ssum,
                          const float* __restrict__ h, int dout,
                          float* feat, int ldf, int coff, int total) {
    for (int gid = blockIdx.x * blockDim.x + threadIdx.x; gid < total;
         gid += gridDim.x * blockDim.x) {
        int e = gid / dout, k = gid % dout;
        int ds = (e < E_EDGES) ? ei[E_EDGES + e] : e - E_EDGES;
        int sr = (e < E_EDGES) ? ei[e] : e - E_EDGES;
        float w = evals[e] / ssum[ds];
        atomicAdd(&feat[ds * ldf + coff + k], h[sr * dout + k] * w);
    }
}

// ---------------------------------------------------------------------------
// fused pairwise-distance + top-30 selection (never materializes 10000x10000)
// one block per row: per-thread sorted lists in LDS, then parallel 30-way merge
// ---------------------------------------------------------------------------
#define KT 128
__global__ void k_knn(const float* __restrict__ X1, const float* __restrict__ X2,
                      int* __restrict__ ni) {
    __shared__ float ld[KT * K_NEAR];
    __shared__ int   li[KT * K_NEAR];
    __shared__ int   cnt[KT];
    __shared__ int   head[KT];
    __shared__ float rmin[KT];
    __shared__ int   rarg[KT];
    __shared__ int   sel[K_NEAR];

    int i = blockIdx.x, t = threadIdx.x;
    float ax = X1[i * 10 + 0], ay = X1[i * 10 + 1], az = X1[i * 10 + 2];
    float a2 = ax * ax + ay * ay + az * az;

    float* md = ld + t * K_NEAR;
    int*   mi = li + t * K_NEAR;
    int c = 0;
    for (int j = t; j < N_NODES; j += KT) {
        float bx = X2[j * 10 + 0], by = X2[j * 10 + 1], bz = X2[j * 10 + 2];
        float d2 = a2 + (bx * bx + by * by + bz * bz) -
                   2.f * (ax * bx + ay * by + az * bz);
        if (c < K_NEAR) {
            int p = c++;
            while (p > 0 && md[p - 1] > d2) { md[p] = md[p - 1]; mi[p] = mi[p - 1]; --p; }
            md[p] = d2; mi[p] = j;
        } else if (d2 < md[K_NEAR - 1]) {
            int p = K_NEAR - 1;
            while (p > 0 && md[p - 1] > d2) { md[p] = md[p - 1]; mi[p] = mi[p - 1]; --p; }
            md[p] = d2; mi[p] = j;
        }
    }
    cnt[t] = c;
    head[t] = 0;
    __syncthreads();

    for (int r = 0; r < K_NEAR; ++r) {
        rmin[t] = (head[t] < cnt[t]) ? md[head[t]] : INFINITY;
        rarg[t] = t;
        __syncthreads();
        for (int s = KT / 2; s > 0; s >>= 1) {
            if (t < s && rmin[t + s] < rmin[t]) { rmin[t] = rmin[t + s]; rarg[t] = rarg[t + s]; }
            __syncthreads();
        }
        if (t == 0) {
            int w = rarg[0];
            sel[r] = li[w * K_NEAR + head[w]];
            head[w]++;
        }
        __syncthreads();
    }
    if (t == 0) {
        for (int a = 1; a < K_NEAR; ++a) {  // sort indices ascending (jnp.sort)
            int v = sel[a], p = a;
            while (p > 0 && sel[p - 1] > v) { sel[p] = sel[p - 1]; --p; }
            sel[p] = v;
        }
        for (int r = 0; r < K_NEAR; ++r) ni[i * K_NEAR + r] = sel[r];
    }
}

// ---------------------------------------------------------------------------
// per-pair MLP head: diff -> relu outer MLP -> 10 -> 2
// ---------------------------------------------------------------------------
__global__ void k_pair(const float* __restrict__ pred1, const float* __restrict__ pred2,
                       const int* __restrict__ ni,
                       const float* __restrict__ w1, const float* __restrict__ b1,
                       const float* __restrict__ w2, const float* __restrict__ b2p,
                       const float* __restrict__ mw, const float* __restrict__ mb,
                       const float* __restrict__ m2w, const float* __restrict__ m2b,
                       float* __restrict__ out) {
    int p = blockIdx.x * blockDim.x + threadIdx.x;
    if (p >= N_NODES * K_NEAR) return;
    int i = p / K_NEAR;
    int j = ni[p];

    float diff[32];
#pragma unroll
    for (int k = 0; k < 32; ++k) diff[k] = fabsf(pred1[i * 32 + k] - pred2[j * 32 + k]);

    float b2 = b2p[0];
    float s[32];
    for (int k = 0; k < 32; ++k) {
        float acc = 0.f;
#pragma unroll
        for (int q = 0; q < 32; ++q) {
            float hv = diff[k] * w1[q] + b1[q];
            acc += fmaxf(hv, 0.f) * w2[q];
        }
        s[k] = acc + b2;
    }
    float tv[10];
    for (int r = 0; r < 10; ++r) {
        float acc = mb[r];
#pragma unroll
        for (int k = 0; k < 32; ++k) acc += s[k] * mw[r * 32 + k];
        tv[r] = acc;
    }
    for (int o = 0; o < 2; ++o) {
        float acc = m2b[o];
#pragma unroll
        for (int r = 0; r < 10; ++r) acc += tv[r] * m2w[o * 10 + r];
        out[p * 2 + o] = acc;
    }
}

// ---------------------------------------------------------------------------
// host orchestration
// ---------------------------------------------------------------------------
struct ConvP {
    const float *w, *as, *ad, *b, *we, *ae;
    int din, dout;
};

static void gemm_f16(const float* A, int lda, int offA, int K,
                     const float* W, const float* bias, int Ncols,
                     float* C, int ldc, int offC,
                     _Float16* Ah, _Float16* Wh, hipStream_t stream) {
    int Kpad = (K + 31) & ~31;
    int Npad = (Ncols + 15) & ~15;
    k_cvt_a<<<(N_NODES * Kpad + 255) / 256, 256, 0, stream>>>(A, lda, offA, K, Kpad, Ah);
    k_cvt_w<<<(Npad * Kpad + 255) / 256, 256, 0, stream>>>(W, Ncols, K, Npad, Kpad, Wh);
    dim3 gg(N_NODES / 16, Npad / 16);
    k_gemm_wmma<<<gg, 32, 0, stream>>>(Ah, Wh, bias, C, ldc, offC, Ncols, Kpad);
}

static void run_conv(const ConvP& c, int coffIn, int coffOut, int ldf,
                     const int* ei, const float* eattr, const float* eamean, float* ve,
                     float* feat, float* h, float* alphaB, float* evals,
                     float* asrc, float* adst, float* amax, float* ssum,
                     _Float16* Ah, _Float16* Wh, hipStream_t stream) {
    const int N = N_NODES, M = E_EDGES + N_NODES;
    gemm_f16(feat, ldf, coffIn, c.din, c.w, nullptr, c.dout, h, c.dout, 0, Ah, Wh, stream);
    k_scores<<<(N + 255) / 256, 256, 0, stream>>>(h, c.as, c.ad, asrc, adst, c.dout);
    k_fill<<<(N + 255) / 256, 256, 0, stream>>>(amax, -INFINITY, N);
    k_fill<<<(N + 255) / 256, 256, 0, stream>>>(ssum, 0.f, N);
    k_bias_init<<<(N * c.dout + 255) / 256, 256, 0, stream>>>(feat, c.b, c.dout, ldf, coffOut);
    int hasEA = (eattr != nullptr) ? 1 : 0;
    if (hasEA) k_ve<<<1, 32, 0, stream>>>(c.we, c.ae, ve, c.dout);
    k_alpha<<<(M + 255) / 256, 256, 0, stream>>>(ei, asrc, adst, eattr, ve, eamean,
                                                 1.f / (float)E_EDGES, hasEA, alphaB, amax);
    k_exp<<<(M + 255) / 256, 256, 0, stream>>>(ei, alphaB, amax, evals, ssum);
    int total = M * c.dout;
    int blocks = (total + 255) / 256;
    if (blocks > 8192) blocks = 8192;
    k_scatter<<<blocks, 256, 0, stream>>>(ei, evals, ssum, h, c.dout, feat, ldf, coffOut, total);
}

extern "C" void kernel_launch(void* const* d_in, const int* in_sizes, int n_in,
                              void* d_out, int out_size, void* d_ws, size_t ws_size,
                              hipStream_t stream) {
    (void)in_sizes; (void)n_in; (void)out_size; (void)ws_size;

    const float* x1  = (const float*)d_in[0];
    const int*   ei1 = (const int*)  d_in[1];
    const float* ea1 = (const float*)d_in[2];
    const float* x2  = (const float*)d_in[3];
    const int*   ei2 = (const int*)  d_in[4];
    const float* ea2 = (const float*)d_in[5];

    int P = 6;
    ConvP e1c[6];
    const int e1din[6]  = {10, 64, 64, 64, 64, 64};
    const int e1dout[6] = {64, 64, 64, 64, 64, 1};
    for (int i = 0; i < 6; ++i) {
        e1c[i].w  = (const float*)d_in[P++];
        e1c[i].as = (const float*)d_in[P++];
        e1c[i].ad = (const float*)d_in[P++];
        e1c[i].b  = (const float*)d_in[P++];
        e1c[i].we = (const float*)d_in[P++];
        e1c[i].ae = (const float*)d_in[P++];
        e1c[i].din = e1din[i]; e1c[i].dout = e1dout[i];
    }
    const float* c1w1 = (const float*)d_in[P++];
    const float* c1b1 = (const float*)d_in[P++];

    ConvP e2c[4];
    const int e2din[4]  = {10, 8, 8, 8};
    const int e2dout[4] = {8, 8, 8, 1};
    for (int i = 0; i < 4; ++i) {
        e2c[i].w  = (const float*)d_in[P++];
        e2c[i].as = (const float*)d_in[P++];
        e2c[i].ad = (const float*)d_in[P++];
        e2c[i].b  = (const float*)d_in[P++];
        e2c[i].we = nullptr; e2c[i].ae = nullptr;
        e2c[i].din = e2din[i]; e2c[i].dout = e2dout[i];
    }
    const float* c1w2 = (const float*)d_in[P++];
    const float* c1b2 = (const float*)d_in[P++];

    const float* mlp_w1 = (const float*)d_in[P++];
    const float* mlp_b1 = (const float*)d_in[P++];
    const float* mlp_w2 = (const float*)d_in[P++];
    const float* mlp_b2 = (const float*)d_in[P++];
    const float* m_w    = (const float*)d_in[P++];
    const float* m_b    = (const float*)d_in[P++];
    const float* m2_w   = (const float*)d_in[P++];
    const float* m2_b   = (const float*)d_in[P++];

    // workspace bump allocator
    char* ws = (char*)d_ws;
    size_t off = 0;
    auto alloc = [&](size_t bytes) -> void* {
        void* p = ws + off;
        off += (bytes + 255) & ~(size_t)255;
        return p;
    };
    const int N = N_NODES, M = E_EDGES + N_NODES;
    float*    feat   = (float*)   alloc(sizeof(float) * (size_t)N * 331);
    float*    h      = (float*)   alloc(sizeof(float) * (size_t)N * 64);
    float*    alphaB = (float*)   alloc(sizeof(float) * M);
    float*    evals  = (float*)   alloc(sizeof(float) * M);
    float*    asrc   = (float*)   alloc(sizeof(float) * N);
    float*    adst   = (float*)   alloc(sizeof(float) * N);
    float*    amax   = (float*)   alloc(sizeof(float) * N);
    float*    ssum   = (float*)   alloc(sizeof(float) * N);
    float*    pred1  = (float*)   alloc(sizeof(float) * (size_t)N * 32);
    float*    pred2  = (float*)   alloc(sizeof(float) * (size_t)N * 32);
    int*      ni     = (int*)     alloc(sizeof(int) * (size_t)N * K_NEAR);
    float*    eamean = (float*)   alloc(sizeof(float) * 4);
    float*    ve     = (float*)   alloc(sizeof(float) * 4);
    _Float16* Ah     = (_Float16*)alloc(sizeof(_Float16) * (size_t)N * 352);  // Kpad max 352
    _Float16* Wh     = (_Float16*)alloc(sizeof(_Float16) * (size_t)64 * 352);

    for (int g = 0; g < 2; ++g) {
        const float* x  = g ? x2 : x1;
        const int*   ei = g ? ei2 : ei1;
        const float* ea = g ? ea2 : ea1;
        float* pred     = g ? pred2 : pred1;

        // ---- enc2 (no edge attrs, hidden 8). Only conv0/conv1 feed kw=25.
        {
            const int LDF = 35;
            k_copy_x<<<(N * 10 + 255) / 256, 256, 0, stream>>>(feat, x, LDF);
            run_conv(e2c[0], 0, 10, LDF, ei, nullptr, eamean, ve,
                     feat, h, alphaB, evals, asrc, adst, amax, ssum, Ah, Wh, stream);
            run_conv(e2c[1], 10, 18, LDF, ei, nullptr, eamean, ve,
                     feat, h, alphaB, evals, asrc, adst, amax, ssum, Ah, Wh, stream);
            gemm_f16(feat, LDF, 0, 25, c1w2, c1b2, 16, pred, 32, 0, Ah, Wh, stream);
        }
        // ---- enc1 (edge attrs, hidden 64). conv0..conv4 feed kw=321; conv5 dead.
        {
            const int LDF = 331;
            k_copy_x<<<(N * 10 + 255) / 256, 256, 0, stream>>>(feat, x, LDF);
            k_fill<<<1, 32, 0, stream>>>(eamean, 0.f, 4);
            k_ea_accum<<<256, 256, 0, stream>>>(ea, eamean);
            int coffIn = 0, coffOut = 10;
            for (int l = 0; l < 5; ++l) {
                run_conv(e1c[l], coffIn, coffOut, LDF, ei, ea, eamean, ve,
                         feat, h, alphaB, evals, asrc, adst, amax, ssum, Ah, Wh, stream);
                coffIn = coffOut;
                coffOut += e1c[l].dout;
            }
            gemm_f16(feat, LDF, 0, 321, c1w1, c1b1, 16, pred, 32, 16, Ah, Wh, stream);
        }
    }

    // ---- fused kNN top-30 per row (no 400 MB distance matrix)
    k_knn<<<N_NODES, KT, 0, stream>>>(x1, x2, ni);

    // ---- per-pair MLP head -> output (N1*K_NEAR, 2)
    const int PAIRS = N_NODES * K_NEAR;
    k_pair<<<(PAIRS + 255) / 256, 256, 0, stream>>>(pred1, pred2, ni,
                                                    mlp_w1, mlp_b1, mlp_w2, mlp_b2,
                                                    m_w, m_b, m2_w, m2_b,
                                                    (float*)d_out);
}
